// causalMultiHeadSelfAttention_13039520711116
// MI455X (gfx1250) — matrix operand order repaired and checked by reading, not compiled, not run on hardware
//
#include <hip/hip_runtime.h>

typedef __attribute__((ext_vector_type(16))) __bf16 v16bf;
typedef __attribute__((ext_vector_type(8)))  __bf16 v8bf;
typedef __attribute__((ext_vector_type(8)))  float  v8f;

#define B_   2
#define S_   2048
#define D_   1024
#define H_   16
#define DK_  64
#define BS_  (B_ * S_)

__device__ __forceinline__ unsigned short f2bf(float f) {
  union { float f; unsigned u; } v; v.f = f;
  unsigned r = v.u + 0x7FFFu + ((v.u >> 16) & 1u);   // round-to-nearest-even
  return (unsigned short)(r >> 16);
}

__device__ __forceinline__ v8f vzero8() {
  v8f z;
#pragma unroll
  for (int i = 0; i < 8; i++) z[i] = 0.0f;
  return z;
}

// A-operand: two 16B chunks (K c0..c0+7 and c0+16..c0+23)
__device__ __forceinline__ v16bf ld_a16(const unsigned short* p0, const unsigned short* p1) {
  v8bf lo = *(const v8bf*)p0;
  v8bf hi = *(const v8bf*)p1;
  return __builtin_shufflevector(lo, hi, 0,1,2,3,4,5,6,7,8,9,10,11,12,13,14,15);
}
// B-operand: two 16B chunks of one column (row of B^T): K c0..c0+7 and c0+16..c0+23
__device__ __forceinline__ v16bf ld_b16(const unsigned short* p) {
  v8bf lo = *(const v8bf*)p;
  v8bf hi = *(const v8bf*)(p + 16);
  return __builtin_shufflevector(lo, hi, 0,1,2,3,4,5,6,7,8,9,10,11,12,13,14,15);
}

// CDNA5 async DMA: memory -> LDS, 16B per lane, tracked by ASYNCcnt.
// lds_off: raw LDS byte address (low 32 bits of generic shared pointer).
__device__ __forceinline__ void async_ld_b128(unsigned lds_off, const void* gptr) {
  unsigned long long ga = (unsigned long long)(size_t)gptr;
  asm volatile("global_load_async_to_lds_b128 %0, %1, off"
               :: "v"(lds_off), "v"(ga)
               : "memory");
}
__device__ __forceinline__ void wait_async0() {
  asm volatile("s_wait_asynccnt 0x0" ::: "memory");
}
__device__ __forceinline__ unsigned lds_off32(const void* p) {
  return (unsigned)(size_t)p;   // LDS aperture: addr[31:0] == LDS byte offset
}

// ---------------- converts ----------------
__global__ void k_cvt(const float* __restrict__ in, unsigned short* __restrict__ out, int n) {
  int i = blockIdx.x * blockDim.x + threadIdx.x;
  if (i < n) out[i] = f2bf(in[i]);
}

__global__ void k_cvt_t(const float* __restrict__ in, unsigned short* __restrict__ out) {
  int i = blockIdx.x * blockDim.x + threadIdx.x;   // over D*D
  if (i >= D_ * D_) return;
  int c = i >> 10, j = i & 1023;
  out[(size_t)j * D_ + c] = f2bf(in[i]);           // Wo^T
}

// ---------------- RoPE (interleaved pairs) + bf16 ----------------
__global__ void k_rope(const float* __restrict__ inF, const int* __restrict__ pos,
                       unsigned short* __restrict__ outB) {
  int idx = blockIdx.x * blockDim.x + threadIdx.x;   // B*H*S*32 pairs
  if (idx >= B_ * H_ * S_ * (DK_ / 2)) return;
  int i  = idx & 31;
  int s  = (idx >> 5) & (S_ - 1);
  int bh = idx >> 16;
  size_t base = ((size_t)bh * S_ + s) * DK_ + 2 * i;
  float x1 = inF[base], x2 = inF[base + 1];
  float p  = (float)pos[s];
  float fr = __powf(1000.0f, -(float)(2 * i) / 64.0f);
  float sn, cs;
  __sincosf(p * fr, &sn, &cs);
  outB[base]     = f2bf(x1 * cs - x2 * sn);
  outB[base + 1] = f2bf(x1 * sn + x2 * cs);
}

// ---------------- generic bf16 GEMM: C[M,N] = A[MxK] * Bt[NxK]^T ----------------
// EPI 0: Cf row-major [M,N]
// EPI 1: scatter f32 to [B,H,S,dk]   (n -> h = n&15, k = n>>4)
// EPI 2: scatter bf16 to [B,H,dk,S]  (V transposed, B-operand ready)
template <int EPI>
__global__ __launch_bounds__(256) void k_gemm_bf16(
    const unsigned short* __restrict__ A, const unsigned short* __restrict__ Bt,
    int M, int N, int K,
    float* __restrict__ Cf, unsigned short* __restrict__ Cb) {
  __shared__ __align__(32) unsigned short As[128][32];
  __shared__ __align__(32) unsigned short Bs[128][32];

  const int tid  = threadIdx.x;
  const int lane = tid & 31;
  const int wid  = tid >> 5;
  const int waveM = wid >> 2;   // 0..1  (64 rows each)
  const int waveN = wid & 3;    // 0..3  (32 cols each)
  const int bM = blockIdx.y * 128;
  const int bN = blockIdx.x * 128;

  v8f acc[4][2];
#pragma unroll
  for (int i = 0; i < 4; i++)
#pragma unroll
    for (int j = 0; j < 2; j++) acc[i][j] = vzero8();

  const int ldRow = tid >> 1;
  const int ldCol = (tid & 1) * 16;
  const unsigned short* Arow = A + (size_t)(bM + ldRow) * K;
  const unsigned short* Brow = Bt + (size_t)(bN + ldRow) * K;
  const unsigned ldsA = lds_off32(&As[ldRow][ldCol]);
  const unsigned ldsB = lds_off32(&Bs[ldRow][ldCol]);

  const int rsel = lane & 15;
  const int c0   = (lane < 16) ? 0 : 8;    // A K-chunk base
  const int kbB  = (lane < 16) ? 0 : 8;    // B K-chunk base

  for (int kk = 0; kk < K; kk += 32) {
    __syncthreads();                       // all waves done reading old tile
    async_ld_b128(ldsA,      Arow + kk + ldCol);
    async_ld_b128(ldsA + 16, Arow + kk + ldCol + 8);
    async_ld_b128(ldsB,      Brow + kk + ldCol);
    async_ld_b128(ldsB + 16, Brow + kk + ldCol + 8);
    if (kk + 32 < K) {
      __builtin_prefetch(Arow + kk + 32 + ldCol, 0, 1);   // global_prefetch_b8
      __builtin_prefetch(Brow + kk + 32 + ldCol, 0, 1);
    }
    wait_async0();                          // this wave's DMAs landed
    __syncthreads();                        // all waves' DMAs landed

    v16bf a[4], b[2];
#pragma unroll
    for (int mt = 0; mt < 4; mt++) {
      const unsigned short* p = &As[waveM * 64 + mt * 16 + rsel][c0];
      a[mt] = ld_a16(p, p + 16);
    }
#pragma unroll
    for (int nt = 0; nt < 2; nt++)
      b[nt] = ld_b16(&Bs[waveN * 32 + nt * 16 + rsel][kbB]);

#pragma unroll
    for (int mt = 0; mt < 4; mt++)
#pragma unroll
      for (int nt = 0; nt < 2; nt++)
        acc[mt][nt] = __builtin_amdgcn_wmma_f32_16x16x32_bf16(
            false, a[mt], false, b[nt], (short)0, acc[mt][nt], false, false);
  }

  const int mhi = (lane >> 4) * 8;
#pragma unroll
  for (int mt = 0; mt < 4; mt++)
#pragma unroll
    for (int nt = 0; nt < 2; nt++)
#pragma unroll
      for (int r = 0; r < 8; r++) {
        int gM = bM + waveM * 64 + mt * 16 + mhi + r;
        int gN = bN + waveN * 32 + nt * 16 + rsel;
        float v = acc[mt][nt][r];
        if (EPI == 0) {
          Cf[(size_t)gM * N + gN] = v;
        } else {
          int bb = gM >> 11, s = gM & 2047;
          int h = gN & 15, k = gN >> 4;
          if (EPI == 1)
            Cf[((size_t)(bb * H_ + h) * S_ + s) * DK_ + k] = v;
          else
            Cb[((size_t)(bb * H_ + h) * DK_ + k) * S_ + s] = f2bf(v);
        }
      }
}

// ---------------- flash attention (causal, online softmax) ----------------
__global__ __launch_bounds__(256) void k_flash(
    const unsigned short* __restrict__ Qb, const unsigned short* __restrict__ Kb,
    const unsigned short* __restrict__ Vt, unsigned short* __restrict__ Ob) {
  __shared__ __align__(32) unsigned short Ks[32][64];       // 32 keys x 64 dk
  __shared__ __align__(32) unsigned short Vs[64][32];       // 64 vdim x 32 keys (V^T)
  __shared__ __align__(32) unsigned short Ps[8][16][32];    // per-wave P staging

  const int tid  = threadIdx.x;
  const int lane = tid & 31;
  const int wid  = tid >> 5;
  const int qt = blockIdx.x, h = blockIdx.y, bb = blockIdx.z;

  const size_t headQK = (size_t)(bb * H_ + h) * S_ * DK_;
  const size_t headV  = (size_t)(bb * H_ + h) * DK_ * S_;

  const int qbase = qt * 128 + wid * 16;
  const int rsel  = lane & 15;
  const int c0    = (lane < 16) ? 0 : 8;
  const int kbB   = (lane < 16) ? 0 : 8;

  // Q A-operands, dk split into 2 x K=32, held in registers for all key tiles
  v16bf aq[2];
  {
    const unsigned short* qrow = Qb + headQK + (size_t)(qbase + rsel) * DK_;
#pragma unroll
    for (int ks = 0; ks < 2; ks++) {
      const unsigned short* p = qrow + ks * 32 + c0;
      aq[ks] = ld_a16(p, p + 16);
    }
  }

  v8f o[4];
#pragma unroll
  for (int t = 0; t < 4; t++) o[t] = vzero8();
  float mrun[8], lrun[8];
#pragma unroll
  for (int r = 0; r < 8; r++) { mrun[r] = -1e30f; lrun[r] = 0.0f; }

  // per-thread staging slots (8 ushorts each from K and V^T tiles)
  const int idx8 = tid * 8;
  const int kr = idx8 >> 6, kc = idx8 & 63;
  const int vr = idx8 >> 5, vc = idx8 & 31;
  const unsigned ldsK = lds_off32(&Ks[kr][kc]);
  const unsigned ldsV = lds_off32(&Vs[vr][vc]);

  const int nTiles = (qt + 1) * 4;                    // keys <= end of q tile
  for (int kt = 0; kt < nTiles; kt++) {
    __syncthreads();                                  // done reading old K/V tile
    async_ld_b128(ldsK, Kb + headQK + (size_t)(kt * 32 + kr) * DK_ + kc);
    async_ld_b128(ldsV, Vt + headV + (size_t)vr * S_ + kt * 32 + vc);
    wait_async0();
    __syncthreads();                                  // tile visible to all waves

    // scores: S[16q x 32key] = Q(16x64) * K^T, two key groups of 16
    v8f sc[2];
#pragma unroll
    for (int g = 0; g < 2; g++) {
      v16bf bk0 = ld_b16(&Ks[g * 16 + rsel][kbB]);
      v16bf bk1 = ld_b16(&Ks[g * 16 + rsel][32 + kbB]);
      v8f z = vzero8();
      z = __builtin_amdgcn_wmma_f32_16x16x32_bf16(false, aq[0], false, bk0, (short)0, z, false, false);
      z = __builtin_amdgcn_wmma_f32_16x16x32_bf16(false, aq[1], false, bk1, (short)0, z, false, false);
      sc[g] = z;
    }

    // online softmax per row (rows split across lane halves: M = r + 8*(lane>=16))
#pragma unroll
    for (int r = 0; r < 8; r++) {
      int q = qbase + (lane >> 4) * 8 + r;
      float s0 = sc[0][r] * 0.125f;               // 1/sqrt(64)
      float s1 = sc[1][r] * 0.125f;
      int k0 = kt * 32 + rsel;
      if (k0 > q)       s0 = -1e30f;              // causal mask
      if (k0 + 16 > q)  s1 = -1e30f;
      float lm = fmaxf(s0, s1);
#pragma unroll
      for (int mk = 1; mk < 16; mk <<= 1) lm = fmaxf(lm, __shfl_xor(lm, mk, 32));
      float mnew  = fmaxf(mrun[r], lm);
      float alpha = __expf(mrun[r] - mnew);
      mrun[r] = mnew;
      float p0 = __expf(s0 - mnew);
      float p1 = __expf(s1 - mnew);
      float rs = p0 + p1;
#pragma unroll
      for (int mk = 1; mk < 16; mk <<= 1) rs += __shfl_xor(rs, mk, 32);
      lrun[r] = lrun[r] * alpha + rs;
#pragma unroll
      for (int t = 0; t < 4; t++) o[t][r] *= alpha;
      int Mrow = (lane >> 4) * 8 + r;
      Ps[wid][Mrow][rsel]      = f2bf(p0);        // C-layout -> LDS
      Ps[wid][Mrow][rsel + 16] = f2bf(p1);
    }

    // P(16x32) * V(32x64): reload P in A-layout (same-wave DS is in-order)
    {
      const unsigned short* pp = &Ps[wid][rsel][c0];
      v16bf pa = ld_a16(pp, pp + 16);
#pragma unroll
      for (int t = 0; t < 4; t++) {
        v16bf vb = ld_b16(&Vs[t * 16 + rsel][kbB]);
        o[t] = __builtin_amdgcn_wmma_f32_16x16x32_bf16(
            false, pa, false, vb, (short)0, o[t], false, false);
      }
    }
  }

  // normalize and emit bf16 [B,S,H,dk] (== column order h*64+v for O-proj)
#pragma unroll
  for (int r = 0; r < 8; r++) {
    float inv = 1.0f / lrun[r];
    int q = qbase + (lane >> 4) * 8 + r;
    size_t ob = ((size_t)(bb * S_ + q)) * D_ + h * DK_;
#pragma unroll
    for (int t = 0; t < 4; t++)
      Ob[ob + t * 16 + rsel] = f2bf(o[t][r] * inv);
  }
}

extern "C" void kernel_launch(void* const* d_in, const int* in_sizes, int n_in,
                              void* d_out, int out_size, void* d_ws, size_t ws_size,
                              hipStream_t stream) {
  const float* X  = (const float*)d_in[0];
  const float* Wq = (const float*)d_in[1];
  const float* Wk = (const float*)d_in[2];
  const float* Wv = (const float*)d_in[3];
  const float* Wo = (const float*)d_in[4];
  const int*  pos = (const int*)d_in[5];

  char* ws = (char*)d_ws;
  size_t off = 0;
  auto take = [&](size_t bytes) -> char* {
    char* p = ws + off;
    off += (bytes + 255) & ~(size_t)255;
    return p;
  };
  unsigned short* Xb  = (unsigned short*)take((size_t)BS_ * D_ * 2);
  unsigned short* Wb  = (unsigned short*)take((size_t)3 * D_ * D_ * 2);
  unsigned short* Wot = (unsigned short*)take((size_t)D_ * D_ * 2);
  float*          Qf  = (float*)take((size_t)BS_ * D_ * 4);
  float*          Kf  = (float*)take((size_t)BS_ * D_ * 4);
  unsigned short* Qb2 = (unsigned short*)take((size_t)BS_ * D_ * 2);
  unsigned short* Kb2 = (unsigned short*)take((size_t)BS_ * D_ * 2);
  unsigned short* Vt  = (unsigned short*)take((size_t)BS_ * D_ * 2);
  unsigned short* Ob  = (unsigned short*)take((size_t)BS_ * D_ * 2);

  const int nX = BS_ * D_, nW = D_ * D_;
  k_cvt<<<(nX + 255) / 256, 256, 0, stream>>>(X, Xb, nX);
  k_cvt<<<(nW + 255) / 256, 256, 0, stream>>>(Wq, Wb + 0 * (size_t)nW, nW);
  k_cvt<<<(nW + 255) / 256, 256, 0, stream>>>(Wk, Wb + 1 * (size_t)nW, nW);
  k_cvt<<<(nW + 255) / 256, 256, 0, stream>>>(Wv, Wb + 2 * (size_t)nW, nW);
  k_cvt_t<<<(nW + 255) / 256, 256, 0, stream>>>(Wo, Wot);

  dim3 gGemm(D_ / 128, BS_ / 128);
  k_gemm_bf16<1><<<gGemm, 256, 0, stream>>>(Xb, Wb + 0 * (size_t)nW, BS_, D_, D_, Qf, nullptr);
  k_gemm_bf16<1><<<gGemm, 256, 0, stream>>>(Xb, Wb + 1 * (size_t)nW, BS_, D_, D_, Kf, nullptr);
  k_gemm_bf16<2><<<gGemm, 256, 0, stream>>>(Xb, Wb + 2 * (size_t)nW, BS_, D_, D_, nullptr, Vt);

  const int nR = B_ * H_ * S_ * (DK_ / 2);
  k_rope<<<(nR + 255) / 256, 256, 0, stream>>>(Qf, pos, Qb2);
  k_rope<<<(nR + 255) / 256, 256, 0, stream>>>(Kf, pos, Kb2);

  k_flash<<<dim3(S_ / 128, H_, B_), 256, 0, stream>>>(Qb2, Kb2, Vt, Ob);

  k_gemm_bf16<0><<<gGemm, 256, 0, stream>>>(Ob, Wot, BS_, D_, D_, (float*)d_out, nullptr);
}
